// MSA_44178033607242
// MI455X (gfx1250) — matrix operand order repaired and checked by reading, not compiled, not run on hardware
//
#include <hip/hip_runtime.h>

// ---------------- problem constants ----------------
#define BB 8
#define SS 1024
#define DD 1024
#define HH 16
#define DH 64

typedef __attribute__((ext_vector_type(16))) _Float16 v16h;
typedef __attribute__((ext_vector_type(8)))  _Float16 v8h;
typedef __attribute__((ext_vector_type(8)))  float    v8f;
typedef __attribute__((ext_vector_type(4)))  unsigned int v4u;
typedef __attribute__((ext_vector_type(8)))  int      v8i;
typedef __attribute__((ext_vector_type(4)))  int      v4i;

// D = A(16x32 f16) x B(32x16 f16) + C(16x16 f32)
__device__ inline v8f wmma_f16(v16h a, v16h b, v8f c) {
    return __builtin_amdgcn_wmma_f32_16x16x32_f16(
        /*neg_a=*/false, a, /*neg_b=*/false, b,
        /*c_mod=*/(short)0, c, /*reuse_a=*/false, /*reuse_b=*/false);
}

__device__ inline v16h cat8(v8h lo, v8h hi) {
    return __builtin_shufflevector(lo, hi, 0,1,2,3,4,5,6,7,8,9,10,11,12,13,14,15);
}

// ---------------------------------------------------------------------
// TDM: 2D tile DMA  global -> LDS  (descriptor per CDNA5 ISA)
//   tensor_d0 : line length (elements, f16), bounds x
//   tensor_d1 : number of lines, bounds y
//   stride0   : elements between consecutive lines
//   tile_d0/1 : tile extent (x elements, y lines); LDS layout [tile_d1][tile_d0]
// ---------------------------------------------------------------------
__device__ inline void tdm_load_2d_f16(unsigned lds_addr, const void* gaddr,
                                       unsigned tensor_d0, unsigned tensor_d1,
                                       unsigned long long stride0,
                                       unsigned tile_d0, unsigned tile_d1) {
    unsigned long long ga = (unsigned long long)(uintptr_t)gaddr;
    v4u g0;
    g0.x = 1u;                                         // count=1, user descriptor
    g0.y = lds_addr;                                   // [63:32] lds_addr (bytes)
    g0.z = (unsigned)(ga & 0xFFFFFFFFu);               // [95:64] global_addr[31:0]
    g0.w = (unsigned)((ga >> 32) & 0x1FFFFFFu)         // [120:96] global_addr[56:32]
         | (2u << 30);                                 // [127:126] type=2 (image)
    v8i g1;
    g1[0] = (int)(1u << 16);                           // data_size=1 -> 2 bytes/elem
    g1[1] = (int)((tensor_d0 & 0xFFFFu) << 16);        // tensor_dim0[15:0] @ bits 63:48
    g1[2] = (int)(((tensor_d0 >> 16) & 0xFFFFu)        // tensor_dim0[31:16]
         | ((tensor_d1 & 0xFFFFu) << 16));             // tensor_dim1[15:0]
    g1[3] = (int)(((tensor_d1 >> 16) & 0xFFFFu)        // tensor_dim1[31:16]
         | ((tile_d0 & 0xFFFFu) << 16));               // tile_dim0
    g1[4] = (int)(tile_d1 & 0xFFFFu);                  // tile_dim1 (tile_dim2=0)
    g1[5] = (int)(unsigned)(stride0 & 0xFFFFFFFFull);  // tensor_dim0_stride[31:0]
    g1[6] = (int)(unsigned)((stride0 >> 32) & 0xFFFFull); // stride[47:32], dim1_stride=0
    g1[7] = 0;
    v4i z4 = {0, 0, 0, 0};
#if defined(__clang_major__) && (__clang_major__ >= 23)
    v8i z8 = {0, 0, 0, 0, 0, 0, 0, 0};
    __builtin_amdgcn_tensor_load_to_lds(g0, g1, z4, z4, z8, 0);
#else
    __builtin_amdgcn_tensor_load_to_lds(g0, g1, z4, z4, 0);
#endif
}

// =====================================================================
// Kernel 1: per-head QKV projection.  q = x @ W^T + b  (per head)
// Q,K stored [B,H,S,DH] f16 ; V stored transposed [B,H,DH,S] f16
// One wave = one 16-token strip of one (b,h).
// =====================================================================
__device__ inline void proj_compute(const v16h a[2], const float* __restrict__ W,
                                    const float* __restrict__ bias,
                                    int h, int ln, int hi, v8f c[4]) {
#pragma unroll
    for (int nt = 0; nt < 4; ++nt) {
        v8f acc = {};
#pragma unroll
        for (int kb = 0; kb < 2; ++kb) {
            // B(k=d, n=o) = W[o, d]: lane n holds column o, 16 contiguous d's
            int o  = nt * 16 + ln;
            int d0 = kb * 32 + hi * 16;
            const float* wr = W + ((size_t)(h * DH + o)) * DH + d0;
            v8f w0 = *(const v8f*)(wr);
            v8f w1 = *(const v8f*)(wr + 8);
            v16h bf;
#pragma unroll
            for (int i = 0; i < 8; ++i) { bf[i] = (_Float16)w0[i]; bf[i + 8] = (_Float16)w1[i]; }
            acc = wmma_f16(a[kb], bf, acc);
        }
        float bv = bias[h * DH + nt * 16 + ln];   // C layout: N = ln for all rows
#pragma unroll
        for (int r = 0; r < 8; ++r) acc[r] += bv;
        c[nt] = acc;
    }
}

__global__ __launch_bounds__(256) void qkv_proj(
    const float* __restrict__ x,
    const float* __restrict__ Wq, const float* __restrict__ bq,
    const float* __restrict__ Wk, const float* __restrict__ bk,
    const float* __restrict__ Wv, const float* __restrict__ bv,
    _Float16* __restrict__ Q, _Float16* __restrict__ K, _Float16* __restrict__ Vt) {

    const int lane  = threadIdx.x & 31;
    const int wave  = threadIdx.x >> 5;
    const int strip = blockIdx.x * 8 + wave;     // 0..8191
    const int bh    = strip >> 6;                // (b*H + h)
    const int s16   = strip & 63;                // 16-row strip within S
    const int h     = bh & (HH - 1);
    const int b     = bh >> 4;
    const int ln    = lane & 15;
    const int hi    = lane >> 4;                 // half-wave select

    // A fragments of x (f32 -> f16): lanes 0-15 row=ln K={0..15}; 16-31 K={16..31}
    const float* xr = x + ((size_t)(b * SS + s16 * 16 + ln)) * DD + h * DH;
    v16h a[2];
#pragma unroll
    for (int kb = 0; kb < 2; ++kb) {
        int d0 = kb * 32 + hi * 16;
        v8f g0 = *(const v8f*)(xr + d0);
        v8f g1 = *(const v8f*)(xr + d0 + 8);
        v16h t;
#pragma unroll
        for (int i = 0; i < 8; ++i) { t[i] = (_Float16)g0[i]; t[i + 8] = (_Float16)g1[i]; }
        a[kb] = t;
    }

    v8f c[4];

    proj_compute(a, Wq, bq, h, ln, hi, c);
#pragma unroll
    for (int nt = 0; nt < 4; ++nt)
#pragma unroll
        for (int r = 0; r < 8; ++r) {
            int srow = s16 * 16 + hi * 8 + r;
            Q[((size_t)bh * SS + srow) * DH + nt * 16 + ln] = (_Float16)c[nt][r];
        }

    proj_compute(a, Wk, bk, h, ln, hi, c);
#pragma unroll
    for (int nt = 0; nt < 4; ++nt)
#pragma unroll
        for (int r = 0; r < 8; ++r) {
            int srow = s16 * 16 + hi * 8 + r;
            K[((size_t)bh * SS + srow) * DH + nt * 16 + ln] = (_Float16)c[nt][r];
        }

    proj_compute(a, Wv, bv, h, ln, hi, c);
#pragma unroll
    for (int nt = 0; nt < 4; ++nt) {
        v8h pk;
#pragma unroll
        for (int r = 0; r < 8; ++r) pk[r] = (_Float16)c[nt][r];
        int d = nt * 16 + ln;
        *(v8h*)(Vt + ((size_t)bh * DH + d) * SS + s16 * 16 + hi * 8) = pk;
    }
}

// =====================================================================
// Kernel 2: flash attention per (b,h); 128 threads = 4 waves, each wave
// owns 16 query rows.  K/V tiles (64 keys) are DMA'd into LDS by the TDM
// (double-buffered), all waves consume them via ds_load fragments.
// =====================================================================
__global__ __launch_bounds__(128) void attn(
    const _Float16* __restrict__ Q, const _Float16* __restrict__ K,
    const _Float16* __restrict__ Vt, float* __restrict__ out) {

    __shared__ _Float16 Kt[2][64 * 64];    // [key][d]   8KB x2
    __shared__ _Float16 Vs[2][64 * 64];    // [d][key]   8KB x2
    __shared__ _Float16 plds[4][16 * 64];  // per-wave P restage, 2KB x4

    const int lane = threadIdx.x & 31;
    const int wave = threadIdx.x >> 5;
    const int bh   = blockIdx.x >> 4;       // (b*H + h)
    const int qb   = blockIdx.x & 15;       // 64-row query block
    const int b    = bh >> 4;
    const int h    = bh & (HH - 1);
    const int ln   = lane & 15;
    const int hi   = lane >> 4;
    const int qbase = qb * 64 + wave * 16;

    const _Float16* Qb = Q  + (size_t)bh * SS * DH;
    const _Float16* Kb = K  + (size_t)bh * SS * DH;
    const _Float16* Vb = Vt + (size_t)bh * DH * SS;

    // ---- Q A-fragments (16 rows x 64 d = two k=32 fragments) ----
    v16h qa[2];
    {
        const _Float16* qr = Qb + (size_t)(qbase + ln) * DH;
#pragma unroll
        for (int kb = 0; kb < 2; ++kb) {
            int d0 = kb * 32 + hi * 16;
            v8h g0 = *(const v8h*)(qr + d0);
            v8h g1 = *(const v8h*)(qr + d0 + 8);
            qa[kb] = cat8(g0, g1);
        }
    }

    v8f o[4] = {v8f{}, v8f{}, v8f{}, v8f{}};
    float mrow[8], lrow[8];
#pragma unroll
    for (int r = 0; r < 8; ++r) { mrow[r] = -1.0e30f; lrow[r] = 0.0f; }

    _Float16* myp = plds[wave];

    // ---- TDM prologue: load first 64-key K and V^T tiles ----
    if (wave == 0) {
        tdm_load_2d_f16((unsigned)(uintptr_t)&Kt[0][0], Kb,      DH, 64,   DH, DH, 64);
        tdm_load_2d_f16((unsigned)(uintptr_t)&Vs[0][0], Vb,      64, DH,   SS, 64, DH);
        __builtin_amdgcn_s_wait_tensorcnt(0);
    }
    __syncthreads();

    for (int kk = 0, it = 0; kk < SS; kk += 64, ++it) {
        const int cur = it & 1;

        // prefetch next tile into the other buffer (safe: it was last read
        // in iteration it-1, fenced by the barrier that ended it)
        if (wave == 0 && kk + 64 < SS) {
            tdm_load_2d_f16((unsigned)(uintptr_t)&Kt[cur ^ 1][0],
                            Kb + (size_t)(kk + 64) * DH, DH, 64, DH, DH, 64);
            tdm_load_2d_f16((unsigned)(uintptr_t)&Vs[cur ^ 1][0],
                            Vb + (kk + 64),              64, DH, SS, 64, DH);
        }

        // ---- S = Q . K^T : 4 column tiles x 2 k-steps (from LDS) ----
        v8f s[4] = {v8f{}, v8f{}, v8f{}, v8f{}};
#pragma unroll
        for (int nt = 0; nt < 4; ++nt) {
            const _Float16* kr = &Kt[cur][(nt * 16 + ln) * DH];
#pragma unroll
            for (int kb = 0; kb < 2; ++kb) {
                v16h kf = *(const v16h*)(kr + kb * 32 + hi * 16);
                s[nt] = wmma_f16(qa[kb], kf, s[nt]);
            }
        }
#pragma unroll
        for (int nt = 0; nt < 4; ++nt) s[nt] = s[nt] * 0.125f;   // 1/sqrt(DH)

        // ---- online softmax over 64 keys (rows across 16-lane halves) ----
        float al[8];
#pragma unroll
        for (int r = 0; r < 8; ++r) {
            float t = fmaxf(fmaxf(s[0][r], s[1][r]), fmaxf(s[2][r], s[3][r]));
            t = fmaxf(t, __shfl_xor(t, 1));
            t = fmaxf(t, __shfl_xor(t, 2));
            t = fmaxf(t, __shfl_xor(t, 4));
            t = fmaxf(t, __shfl_xor(t, 8));
            float mnew = fmaxf(mrow[r], t);
            al[r] = __expf(mrow[r] - mnew);
            float rs = 0.0f;
#pragma unroll
            for (int nt = 0; nt < 4; ++nt) {
                float p = __expf(s[nt][r] - mnew);
                s[nt][r] = p;
                rs += p;
            }
            rs += __shfl_xor(rs, 1);
            rs += __shfl_xor(rs, 2);
            rs += __shfl_xor(rs, 4);
            rs += __shfl_xor(rs, 8);
            lrow[r] = lrow[r] * al[r] + rs;
            mrow[r] = mnew;
        }
#pragma unroll
        for (int nt = 0; nt < 4; ++nt)
#pragma unroll
            for (int r = 0; r < 8; ++r) o[nt][r] *= al[r];

        // ---- restage P through per-wave LDS: C-layout -> row-major [16][64] ----
#pragma unroll
        for (int r = 0; r < 8; ++r) {
            int row = hi * 8 + r;
#pragma unroll
            for (int nt = 0; nt < 4; ++nt)
                myp[row * 64 + nt * 16 + ln] = (_Float16)s[nt][r];
        }
        v16h pa[2];   // A-fragments of P, two k=32 blocks
#pragma unroll
        for (int kb = 0; kb < 2; ++kb) {
            v8h pg0 = *(const v8h*)(myp + ln * 64 + kb * 32 + hi * 16);
            v8h pg1 = *(const v8h*)(myp + ln * 64 + kb * 32 + 8 + hi * 16);
            pa[kb] = cat8(pg0, pg1);
        }

        // ---- O += P . V : 4 d-tiles x 2 k-steps (V^T tile in LDS) ----
#pragma unroll
        for (int nt = 0; nt < 4; ++nt) {
            const _Float16* vr = &Vs[cur][(nt * 16 + ln) * 64];
#pragma unroll
            for (int kb = 0; kb < 2; ++kb) {
                v16h vf = *(const v16h*)(vr + kb * 32 + hi * 16);
                o[nt] = wmma_f16(pa[kb], vf, o[nt]);
            }
        }

        if (wave == 0) __builtin_amdgcn_s_wait_tensorcnt(0);
        __syncthreads();
    }

    // ---- normalize and store fp32 output [B,S,H*DH] ----
    float inv[8];
#pragma unroll
    for (int r = 0; r < 8; ++r) inv[r] = 1.0f / lrow[r];
#pragma unroll
    for (int nt = 0; nt < 4; ++nt)
#pragma unroll
        for (int r = 0; r < 8; ++r) {
            int srow = qbase + hi * 8 + r;
            out[((size_t)(b * SS + srow)) * DD + h * DH + nt * 16 + ln] = o[nt][r] * inv[r];
        }
}

// =====================================================================
extern "C" void kernel_launch(void* const* d_in, const int* in_sizes, int n_in,
                              void* d_out, int out_size, void* d_ws, size_t ws_size,
                              hipStream_t stream) {
    const float* images = (const float*)d_in[0];
    const float* Wq = (const float*)d_in[1];
    const float* bq = (const float*)d_in[2];
    const float* Wk = (const float*)d_in[3];
    const float* bk = (const float*)d_in[4];
    const float* Wv = (const float*)d_in[5];
    const float* bv = (const float*)d_in[6];
    float* out = (float*)d_out;

    const size_t per = (size_t)BB * HH * SS * DH;   // 8M f16 elements = 16 MB
    _Float16* Qw = (_Float16*)d_ws;
    _Float16* Kw = Qw + per;
    _Float16* Vw = Kw + per;                        // total 48 MB workspace

    qkv_proj<<<1024, 256, 0, stream>>>(images, Wq, bq, Wk, bk, Wv, bv, Qw, Kw, Vw);
    attn<<<2048, 128, 0, stream>>>(Qw, Kw, Vw, out);

    (void)in_sizes; (void)n_in; (void)out_size; (void)ws_size;
}
